// FeedFoward_64854006170068
// MI455X (gfx1250) — compile-verified, matrix operand order checked by reading, not run on hardware
//
#include <hip/hip_runtime.h>
#include <hip/hip_bf16.h>

// ---------------------------------------------------------------------------
// BitNet-style FFN: ternary-quantized weights, f16 WMMA GEMMs, f32 accumulate.
//   h   = relu(x @ Wq1^T + b1)     [16384, 4096]
//   out =      h @ Wq2^T + b2      [16384, 1024]
// Matrix-core bound (275 GFLOP vs ~435 MB traffic). f16 WMMA w/ f32 accum
// (ternary weights exact in f16). Block tile 256x128, wave tile 64x64
// (16 v_wmma / K-step / wave). Double-buffered LDS fed by CDNA5 async
// global->LDS DMA (no VGPR staging -> no spills), 1 barrier per K-step.
// ---------------------------------------------------------------------------

typedef _Float16 v16h __attribute__((ext_vector_type(16)));
typedef _Float16 v8h  __attribute__((ext_vector_type(8)));
typedef _Float16 v4h  __attribute__((ext_vector_type(4)));
typedef float    v8f  __attribute__((ext_vector_type(8)));
typedef int      v4i  __attribute__((ext_vector_type(4)));

#if __has_builtin(__builtin_amdgcn_global_load_async_to_lds_b128) && \
    __has_builtin(__builtin_amdgcn_s_wait_asynccnt)
#define USE_ASYNC_LDS 1
#else
#define USE_ASYNC_LDS 0
#endif

// Builtin signature (from hipcc diagnostic): b128 variant takes int4-vector
// pointers: (v4i addrspace(1)* src, v4i addrspace(3)* dst, Imm off, Imm cpol).
#define AS1V4(p) ((__attribute__((address_space(1))) v4i*)(void*)(p))
#define AS3V4(p) ((__attribute__((address_space(3))) v4i*)(void*)(p))

// ---------------- gamma = mean(|W|), fixed-order two-stage reduction --------

__global__ void absmean_partial(const float* __restrict__ w, int n,
                                float* __restrict__ part) {
    __shared__ float sm[256];
    float s = 0.f;
    for (int i = blockIdx.x * blockDim.x + threadIdx.x; i < n;
         i += gridDim.x * blockDim.x)
        s += fabsf(w[i]);
    sm[threadIdx.x] = s;
    __syncthreads();
    for (int off = 128; off > 0; off >>= 1) {
        if ((int)threadIdx.x < off) sm[threadIdx.x] += sm[threadIdx.x + off];
        __syncthreads();
    }
    if (threadIdx.x == 0) part[blockIdx.x] = sm[0];
}

__global__ void absmean_final(const float* __restrict__ part, int nparts,
                              float invn, float* __restrict__ gamma) {
    __shared__ float sm[256];
    float s = 0.f;
    for (int i = threadIdx.x; i < nparts; i += 256) s += part[i];
    sm[threadIdx.x] = s;
    __syncthreads();
    for (int off = 128; off > 0; off >>= 1) {
        if ((int)threadIdx.x < off) sm[threadIdx.x] += sm[threadIdx.x + off];
        __syncthreads();
    }
    if (threadIdx.x == 0) gamma[0] = sm[0] * invn;
}

// ---------------- ternary quantize to f16 (-1/0/+1 exact in f16) ------------

__global__ void quantize_w(const float* __restrict__ w,
                           const float* __restrict__ gamma,
                           _Float16* __restrict__ wq, int n) {
    int i = blockIdx.x * blockDim.x + threadIdx.x;
    if (i < n) {
        float g = gamma[0] + 1e-5f;
        float q = rintf(w[i] / g);               // round-half-even == jnp.round
        q = fminf(1.f, fmaxf(-1.f, q));
        wq[i] = (_Float16)q;
    }
}

// ---------------- x: f32 -> f16 (vectorized) --------------------------------

__global__ void cvt_f32_to_f16(const float* __restrict__ in,
                               _Float16* __restrict__ out, int n4) {
    int i = blockIdx.x * blockDim.x + threadIdx.x;
    if (i < n4) {
        float4 v = ((const float4*)in)[i];
        v4h o = {(_Float16)v.x, (_Float16)v.y, (_Float16)v.z, (_Float16)v.w};
        ((v4h*)out)[i] = o;
    }
}

// ---------------- tiled WMMA GEMM:  C[M,N] = A[M,K] * B[N,K]^T + bias -------
// Block: 256 threads (8 waves). Block tile 256(M) x 128(N), K-step 32.
// Waves: 4 along M x 2 along N -> each wave owns 64x64 = 4x4 WMMA tiles.

template <bool RELU, bool OUT_F16>
__global__ __launch_bounds__(256)
void gemm_wmma_f16(const _Float16* __restrict__ A,   // [M,K]
                   const _Float16* __restrict__ B,   // [N,K]
                   const float* __restrict__ bias,   // [N]
                   void* __restrict__ Cout,
                   int M, int N, int K) {
#if USE_ASYNC_LDS
    __shared__ _Float16 ldsA[2][256 * 32];           // 2 x 16 KB
    __shared__ _Float16 ldsB[2][128 * 32];           // 2 x  8 KB
#else
    __shared__ _Float16 ldsA[1][256 * 32];
    __shared__ _Float16 ldsB[1][128 * 32];
#endif

    const int tid  = threadIdx.x;
    const int lane = tid & 31;
    const int wave = tid >> 5;
    const int wm   = wave & 3;     // M: rows wm*64 .. +63
    const int wn   = wave >> 2;    // N: cols wn*64 .. +63
    const int m0   = blockIdx.y * 256;
    const int n0   = blockIdx.x * 128;

    // ISA 16-bit A/B fragment addressing (wave32):
    //   lanes 0-15  hold K = 0..7 and 16..23   (kb = 0)
    //   lanes 16-31 hold K = 8..15 and 24..31  (kb = 8)
    const int r  = lane & 15;
    const int kb = (lane >> 4) << 3;

    v8f acc[4][4] = {};

#if USE_ASYNC_LDS
    // DMA one K-tile (A: 1024 b128 segments, B: 512) straight into LDS.
    // Zero VGPR staging; completion tracked by ASYNCcnt.
    auto issue_tile = [&](int buf, int kk) {
#pragma unroll
        for (int s = 0; s < 4; ++s) {
            int seg = tid + s * 256, row = seg >> 2, c8 = (seg & 3) << 3;
            __builtin_amdgcn_global_load_async_to_lds_b128(
                AS1V4(&A[(size_t)(m0 + row) * K + kk + c8]),
                AS3V4(&ldsA[buf][row * 32 + c8]), 0, 0);
        }
#pragma unroll
        for (int s = 0; s < 2; ++s) {
            int seg = tid + s * 256, row = seg >> 2, c8 = (seg & 3) << 3;
            __builtin_amdgcn_global_load_async_to_lds_b128(
                AS1V4(&B[(size_t)(n0 + row) * K + kk + c8]),
                AS3V4(&ldsB[buf][row * 32 + c8]), 0, 0);
        }
    };

    issue_tile(0, 0);                              // prologue: tile 0 -> buf 0
    int cur = 0;
#endif

    for (int k0 = 0; k0 < K; k0 += 32) {
#if USE_ASYNC_LDS
        __builtin_amdgcn_s_wait_asynccnt(0);       // this wave's DMA landed
        __syncthreads();                           // every wave's DMA landed
        if (k0 + 32 < K) {
            issue_tile(cur ^ 1, k0 + 32);          // next tile DMAs under WMMAs
            if (k0 + 64 < K) {                     // warm L2 one tile further
                __builtin_prefetch(&A[(size_t)(m0 + tid) * K + k0 + 64], 0, 3);
                __builtin_prefetch(&B[(size_t)(n0 + (tid >> 1)) * K + k0 + 64], 0, 3);
            }
        }
        const _Float16* la = ldsA[cur];
        const _Float16* lb = ldsB[cur];
        cur ^= 1;
#else
        __syncthreads();
#pragma unroll
        for (int s = 0; s < 4; ++s) {
            int seg = tid + s * 256, row = seg >> 2, c8 = (seg & 3) << 3;
            *(uint4*)&ldsA[0][row * 32 + c8] =
                *(const uint4*)&A[(size_t)(m0 + row) * K + k0 + c8];
        }
#pragma unroll
        for (int s = 0; s < 2; ++s) {
            int seg = tid + s * 256, row = seg >> 2, c8 = (seg & 3) << 3;
            *(uint4*)&ldsB[0][row * 32 + c8] =
                *(const uint4*)&B[(size_t)(n0 + row) * K + k0 + c8];
        }
        __syncthreads();
        const _Float16* la = ldsA[0];
        const _Float16* lb = ldsB[0];
#endif

        v16h afrag[4];
#pragma unroll
        for (int mi = 0; mi < 4; ++mi) {
            int row = wm * 64 + mi * 16 + r;
            v8h lo = *(const v8h*)&la[row * 32 + kb];
            v8h hi = *(const v8h*)&la[row * 32 + kb + 16];
            afrag[mi] = __builtin_shufflevector(lo, hi, 0, 1, 2, 3, 4, 5, 6, 7,
                                                8, 9, 10, 11, 12, 13, 14, 15);
        }
#pragma unroll
        for (int nj = 0; nj < 4; ++nj) {
            int col = wn * 64 + nj * 16 + r;
            v8h lo = *(const v8h*)&lb[col * 32 + kb];
            v8h hi = *(const v8h*)&lb[col * 32 + kb + 16];
            v16h bfrag = __builtin_shufflevector(lo, hi, 0, 1, 2, 3, 4, 5, 6, 7,
                                                 8, 9, 10, 11, 12, 13, 14, 15);
#pragma unroll
            for (int mi = 0; mi < 4; ++mi)
                acc[mi][nj] = __builtin_amdgcn_wmma_f32_16x16x32_f16(
                    /*neg_a=*/false, afrag[mi], /*neg_b=*/false, bfrag,
                    /*c_mod=*/(short)0, acc[mi][nj],
                    /*reuse_a=*/false, /*reuse_b=*/false);
        }
    }

    // Epilogue. C layout: VGPR j -> M = j + 8*(lane>=16), N = lane&15.
    const int rhalf = (lane >> 4) << 3;
#pragma unroll
    for (int mi = 0; mi < 4; ++mi) {
#pragma unroll
        for (int nj = 0; nj < 4; ++nj) {
            int col   = n0 + wn * 64 + nj * 16 + r;
            int rbase = m0 + wm * 64 + mi * 16 + rhalf;
            float bv  = bias[col];
#pragma unroll
            for (int j = 0; j < 8; ++j) {
                float v = acc[mi][nj][j] + bv;
                if (RELU) v = fmaxf(v, 0.f);
                size_t idx = (size_t)(rbase + j) * N + col;
                if (OUT_F16)
                    ((_Float16*)Cout)[idx] = (_Float16)v;
                else
                    ((float*)Cout)[idx] = v;
            }
        }
    }
}

// ---------------------------------------------------------------------------

extern "C" void kernel_launch(void* const* d_in, const int* in_sizes, int n_in,
                              void* d_out, int out_size, void* d_ws,
                              size_t ws_size, hipStream_t stream) {
    const float* x  = (const float*)d_in[0];   // [4,4096,1024]
    const float* W1 = (const float*)d_in[1];   // [4096,1024]
    const float* b1 = (const float*)d_in[2];   // [4096]
    const float* W2 = (const float*)d_in[3];   // [1024,4096]
    const float* b2 = (const float*)d_in[4];   // [1024]
    float* out = (float*)d_out;

    const int n_embd = 1024, d_ff = 4096, Mtok = 16384;
    const size_t nX  = (size_t)Mtok * n_embd;  // 16,777,216
    const size_t nW1 = (size_t)d_ff * n_embd;  //  4,194,304
    const size_t nW2 = (size_t)n_embd * d_ff;  //  4,194,304
    const size_t nH  = (size_t)Mtok * d_ff;    // 67,108,864

    char* ws = (char*)d_ws;
    _Float16* x16 = (_Float16*)ws;  ws += nX * 2;
    _Float16* wq1 = (_Float16*)ws;  ws += nW1 * 2;
    _Float16* wq2 = (_Float16*)ws;  ws += nW2 * 2;
    _Float16* h16 = (_Float16*)ws;  ws += nH * 2;
    float* part1  = (float*)ws;     ws += 512 * sizeof(float);
    float* part2  = (float*)ws;     ws += 512 * sizeof(float);
    float* gamma1 = (float*)ws;     ws += sizeof(float);
    float* gamma2 = (float*)ws;

    // gamma = mean(|W|)
    absmean_partial<<<512, 256, 0, stream>>>(W1, (int)nW1, part1);
    absmean_partial<<<512, 256, 0, stream>>>(W2, (int)nW2, part2);
    absmean_final<<<1, 256, 0, stream>>>(part1, 512, 1.f / (float)nW1, gamma1);
    absmean_final<<<1, 256, 0, stream>>>(part2, 512, 1.f / (float)nW2, gamma2);

    // ternary quantize weights to f16; convert activations to f16
    quantize_w<<<(int)(nW1 / 256), 256, 0, stream>>>(W1, gamma1, wq1, (int)nW1);
    quantize_w<<<(int)(nW2 / 256), 256, 0, stream>>>(W2, gamma2, wq2, (int)nW2);
    cvt_f32_to_f16<<<(int)(nX / 4 / 256), 256, 0, stream>>>(x, x16, (int)(nX / 4));

    // GEMM1: [16384,1024] x [4096,1024]^T -> relu -> h16 [16384,4096]
    gemm_wmma_f16<true, true><<<dim3(d_ff / 128, Mtok / 256), 256, 0, stream>>>(
        x16, wq1, b1, (void*)h16, Mtok, d_ff, n_embd);

    // GEMM2: [16384,4096] x [1024,4096]^T -> out f32 [16384,1024]
    gemm_wmma_f16<false, false><<<dim3(n_embd / 128, Mtok / 256), 256, 0, stream>>>(
        h16, wq2, b2, (void*)out, Mtok, n_embd, d_ff);
}
